// GraphNeuralNetwork_30142080483948
// MI455X (gfx1250) — matrix operand order repaired and checked by reading, not compiled, not run on hardware
//
#include <hip/hip_runtime.h>
#include <hip/hip_bf16.h>

typedef __attribute__((ext_vector_type(16))) __bf16 v16bf;
typedef __attribute__((ext_vector_type(8)))  __bf16 v8bf;
typedef __attribute__((ext_vector_type(8)))  float  v8f;

static inline int cdiv(long a, long b) { return (int)((a + b - 1) / b); }

// ---------------------------------------------------------------- utilities
__global__ void k_zero_f32(float* __restrict__ p, long n) {
    long i = (long)blockIdx.x * blockDim.x + threadIdx.x;
    if (i < n) p[i] = 0.0f;
}

// ---------------------------------------------------------------- degree / norm
__global__ void k_deg(const int* __restrict__ dst, int E, float* __restrict__ deg) {
    long i = (long)blockIdx.x * blockDim.x + threadIdx.x;
    if (i < E) atomicAdd(&deg[dst[i]], 1.0f);
}

__global__ void k_dinv(float* __restrict__ deg, int N) {
    long i = (long)blockIdx.x * blockDim.x + threadIdx.x;
    if (i < N) deg[i] = rsqrtf(deg[i] + 1.0f);   // +1 self-loop; deg>=1 always
}

// ---------------------------------------------------------------- A packing
// f32 row-major [M,Ka] -> bf16 row-major [M,Kpad], zero padded. Off hot path.
__global__ void k_pack_a(const float* __restrict__ A, int Ka, int Kpad,
                         __bf16* __restrict__ out, long M) {
    long i = (long)blockIdx.x * blockDim.x + threadIdx.x;
    long total = M * Kpad;
    if (i >= total) return;
    int  k = (int)(i % Kpad);
    long m = i / Kpad;
    out[i] = (k < Ka) ? (__bf16)A[m * Ka + k] : (__bf16)0.0f;
}

// ---------------------------------------------------------------- B packing
// Packs row-major f32 B[K,N] into bf16 WMMA B-fragments:
//   frag element e of lane l, k-chunk kc, n-tile nt:
//     K = kc*32 + (l>>4)*16 + e,  col = nt*16 + (l&15)
//   stored contiguously: ((kc*nTilesN + nt)*32 + lane)*16 + e
__global__ void k_pack_b(const float* __restrict__ B, int Korig, int N, int Kpad,
                         __bf16* __restrict__ out) {
    long idx = (long)blockIdx.x * blockDim.x + threadIdx.x;
    int nTilesN = N >> 4;
    long total = (long)(Kpad >> 5) * nTilesN * 32 * 16;
    if (idx >= total) return;
    int e    = idx & 15;
    int lane = (idx >> 4) & 31;
    long t   = idx >> 9;
    int nt   = (int)(t % nTilesN);
    int kc   = (int)(t / nTilesN);
    int k    = (kc << 5) + ((lane >> 4) << 4) + e;
    int col  = (nt << 4) + (lane & 15);
    float v  = (k < Korig) ? B[(long)k * N + col] : 0.0f;
    out[idx] = (__bf16)v;
}

// ---------------------------------------------------------------- WMMA GEMM
// C/Cb[M,N] = relu?( A_bf16[M,K](row-major, K%32==0, padded) x Bpacked + bias )
// One 16x(16*NT) output strip per wave; blockDim.x = 256 (8 waves).
template <int K, int NT>
__global__ void k_gemm_wmma(const __bf16* __restrict__ A, int lda,
                            const __bf16* __restrict__ Bp,
                            const float* __restrict__ bias,
                            float* __restrict__ C, __bf16* __restrict__ Cb, int ldc,
                            int M, int N, int relu) {
    int wave = threadIdx.x >> 5;
    int lane = threadIdx.x & 31;
    int nTilesN = N >> 4;
    int nGroups = nTilesN / NT;
    int nTiles  = (M >> 4) * nGroups;
    int tile = blockIdx.x * (blockDim.x >> 5) + wave;   // wave-uniform
    if (tile >= nTiles) return;                          // whole-wave exit: EXEC stays full
    int mt = tile / nGroups;
    int ng = tile % nGroups;

    int rowA = (mt << 4) + (lane & 15);      // A: both half-waves cover M=0..15
    int kgrp = lane >> 4;                    // 0: K 0..15, 1: K 16..31
    const __bf16* Arow = A + (long)rowA * lda;

    v8f zero = {};
    v8f acc[NT];
#pragma unroll
    for (int t = 0; t < NT; ++t) acc[t] = zero;

#pragma unroll 2
    for (int kc = 0; kc < K; kc += 32) {
        int kstart = kc + kgrp * 16;
        v8bf lo = *(const v8bf*)(Arow + kstart);        // 16B contiguous
        v8bf hi = *(const v8bf*)(Arow + kstart + 8);    // 16B contiguous
        v16bf a = __builtin_shufflevector(lo, hi, 0, 1, 2, 3, 4, 5, 6, 7,
                                          8, 9, 10, 11, 12, 13, 14, 15);
#pragma unroll
        for (int t = 0; t < NT; ++t) {
            int nt = ng * NT + t;
            const __bf16* bptr =
                Bp + ((((long)(kc >> 5) * nTilesN + nt) * 32 + lane) << 4);
            v16bf b = *(const v16bf*)bptr;              // 32B contiguous
            acc[t] = __builtin_amdgcn_wmma_f32_16x16x32_bf16(
                false, a, false, b, (short)0, acc[t], false, false);
        }
    }

    int rbase = (mt << 4) + kgrp * 8;        // C: lanes 0-15 -> M 0..7, 16-31 -> M 8..15
#pragma unroll
    for (int t = 0; t < NT; ++t) {
        int col = ((ng * NT + t) << 4) + (lane & 15);
        float bb = bias ? bias[col] : 0.0f;
#pragma unroll
        for (int r = 0; r < 8; ++r) {
            float v = acc[t][r] + bb;
            if (relu) v = fmaxf(v, 0.0f);
            long o = (long)(rbase + r) * ldc + col;
            if (C)  C[o]  = v;
            if (Cb) Cb[o] = (__bf16)v;
        }
    }
}

// Generic runtime-K fallback (NT=1), K must still be a multiple of 32.
__global__ void k_gemm_wmma_gen(const __bf16* __restrict__ A, int lda, int K,
                                const __bf16* __restrict__ Bp,
                                const float* __restrict__ bias,
                                float* __restrict__ C, __bf16* __restrict__ Cb, int ldc,
                                int M, int N, int relu) {
    int wave = threadIdx.x >> 5;
    int lane = threadIdx.x & 31;
    int nTilesN = N >> 4;
    int nTiles  = (M >> 4) * nTilesN;
    int tile = blockIdx.x * (blockDim.x >> 5) + wave;
    if (tile >= nTiles) return;
    int mt = tile / nTilesN;
    int nt = tile % nTilesN;
    int kgrp = lane >> 4;
    const __bf16* Arow = A + (long)((mt << 4) + (lane & 15)) * lda;
    v8f acc = {};
    for (int kc = 0; kc < K; kc += 32) {
        int kstart = kc + kgrp * 16;
        v8bf lo = *(const v8bf*)(Arow + kstart);
        v8bf hi = *(const v8bf*)(Arow + kstart + 8);
        v16bf a = __builtin_shufflevector(lo, hi, 0, 1, 2, 3, 4, 5, 6, 7,
                                          8, 9, 10, 11, 12, 13, 14, 15);
        v16bf b = *(const v16bf*)(Bp + ((((long)(kc >> 5) * nTilesN + nt) * 32 + lane) << 4));
        acc = __builtin_amdgcn_wmma_f32_16x16x32_bf16(
            false, a, false, b, (short)0, acc, false, false);
    }
    int rbase = (mt << 4) + kgrp * 8;
    int col = (nt << 4) + (lane & 15);
    float bb = bias ? bias[col] : 0.0f;
#pragma unroll
    for (int r = 0; r < 8; ++r) {
        float v = acc[r] + bb;
        if (relu) v = fmaxf(v, 0.0f);
        long o = (long)(rbase + r) * ldc + col;
        if (C)  C[o]  = v;
        if (Cb) Cb[o] = (__bf16)v;
    }
}

// ---------------------------------------------------------------- GCN aggregation
// hnew = conv_b + hw * dinv^2  (self-loop term folded into init)
__global__ void k_init_agg(const float* __restrict__ hw, const float* __restrict__ dinv,
                           const float* __restrict__ bias, float* __restrict__ hnew,
                           int Nn, int H) {
    long i = (long)blockIdx.x * blockDim.x + threadIdx.x;
    long total = (long)Nn * H;
    if (i >= total) return;
    int n = (int)(i / H), f = (int)(i % H);
    float d = dinv[n];
    hnew[i] = bias[f] + hw[i] * d * d;
}

__global__ void k_scatter(const int* __restrict__ src, const int* __restrict__ dst,
                          const float* __restrict__ dinv, const float* __restrict__ hw,
                          float* __restrict__ hnew, int E, int H) {
    long tid = (long)blockIdx.x * blockDim.x + threadIdx.x;
    int groups = H >> 2;
    long total = (long)E * groups;
    if (tid >= total) return;
    int e = (int)(tid / groups);
    int g = (int)(tid % groups);
    int s = src[e], d = dst[e];
    float nrm = dinv[s] * dinv[d];
    const float4 v = *(const float4*)(hw + (long)s * H + (g << 2));
    float* outp = hnew + (long)d * H + (g << 2);
    atomicAdd(outp + 0, v.x * nrm);
    atomicAdd(outp + 1, v.y * nrm);
    atomicAdd(outp + 2, v.z * nrm);
    atomicAdd(outp + 3, v.w * nrm);
}

// ---------------------------------------------------------------- BatchNorm
__global__ void k_bn_partial(const float* __restrict__ h, int Nn, int H,
                             float* __restrict__ sums, float* __restrict__ sqs) {
    int f = threadIdx.x % H;                  // blockDim.x = 2*H, coalesced
    int rowsPerBlk = blockDim.x / H;
    int r0 = blockIdx.x * rowsPerBlk + threadIdx.x / H;
    int stride = gridDim.x * rowsPerBlk;
    float s = 0.0f, q = 0.0f;
    for (int n = r0; n < Nn; n += stride) {
        float v = h[(long)n * H + f];
        s += v; q += v * v;
    }
    atomicAdd(&sums[f], s);
    atomicAdd(&sqs[f], q);
}

__global__ void k_bn_finalize(const float* __restrict__ sums, const float* __restrict__ sqs,
                              const float* __restrict__ gamma, const float* __restrict__ beta,
                              float* __restrict__ scale, float* __restrict__ shift,
                              int Nn, int H) {
    int f = threadIdx.x;
    if (f >= H) return;
    float inv = 1.0f / (float)Nn;
    float mu  = sums[f] * inv;
    float var = sqs[f] * inv - mu * mu;       // biased variance (reference)
    float sc  = gamma[f] * rsqrtf(var + 1e-5f);
    scale[f] = sc;
    shift[f] = beta[f] - mu * sc;
}

// h (f32, in place) and hbf (bf16 copy for next GEMM's A operand)
__global__ void k_bn_apply(float* __restrict__ h, __bf16* __restrict__ hbf,
                           const float* __restrict__ scale,
                           const float* __restrict__ shift, int Nn, int H) {
    long i = (long)blockIdx.x * blockDim.x + threadIdx.x;
    long total = (long)Nn * H;
    if (i >= total) return;
    int f = (int)(i % H);
    float v = fmaxf(fmaf(h[i], scale[f], shift[f]), 0.0f);
    h[i]   = v;
    hbf[i] = (__bf16)v;
}

// ---------------------------------------------------------------- pooling
__global__ void k_pool_count(const int* __restrict__ batch, int Nn, float* __restrict__ cnt) {
    long i = (long)blockIdx.x * blockDim.x + threadIdx.x;
    if (i < Nn) atomicAdd(&cnt[batch[i]], 1.0f);
}

__global__ void k_pool_accum(const int* __restrict__ batch, const float* __restrict__ h,
                             int Nn, int H, float* __restrict__ gsum,
                             unsigned int* __restrict__ gmax) {
    long i = (long)blockIdx.x * blockDim.x + threadIdx.x;
    long total = (long)Nn * H;
    if (i >= total) return;
    int n = (int)(i / H), f = (int)(i % H);
    float v = h[i];                            // v >= 0 after ReLU
    long o = (long)batch[n] * H + f;
    atomicAdd(&gsum[o], v);
    atomicMax(&gmax[o], __float_as_uint(v));   // valid order for non-negative floats
}

__global__ void k_build_g(const float* __restrict__ gsum, const unsigned int* __restrict__ gmax,
                          const float* __restrict__ cnt, __bf16* __restrict__ g,
                          int G, int H) {
    long i = (long)blockIdx.x * blockDim.x + threadIdx.x;
    int H2 = 2 * H;
    long total = (long)G * H2;
    if (i >= total) return;
    int gi = (int)(i / H2), f = (int)(i % H2);
    float v;
    if (f < H) v = gsum[(long)gi * H + f] / fmaxf(cnt[gi], 1.0f);
    else       v = __uint_as_float(gmax[(long)gi * H + (f - H)]);
    g[i] = (__bf16)v;
}

// ---------------------------------------------------------------- final N=1 layer
__global__ void k_mlp_out(const float* __restrict__ h2, const float* __restrict__ w3,
                          const float* __restrict__ b3, float* __restrict__ out,
                          int G, int K) {
    long g = (long)blockIdx.x * blockDim.x + threadIdx.x;
    if (g >= G) return;
    float s = b3[0];
    const float* row = h2 + g * K;
    for (int k = 0; k < K; ++k) s = fmaf(row[k], w3[k], s);
    out[g] = s;
}

// ================================================================ host launcher
extern "C" void kernel_launch(void* const* d_in, const int* in_sizes, int n_in,
                              void* d_out, int out_size, void* d_ws, size_t ws_size,
                              hipStream_t stream) {
    const float* x        = (const float*)d_in[0];
    const int*   eidx     = (const int*)  d_in[1];
    const int*   batch    = (const int*)  d_in[2];
    const float* emb_W    = (const float*)d_in[3];
    const float* emb_b    = (const float*)d_in[4];
    const float* conv_W   = (const float*)d_in[5];
    const float* conv_b   = (const float*)d_in[6];
    const float* bn_gamma = (const float*)d_in[7];
    const float* bn_beta  = (const float*)d_in[8];
    const float* w1       = (const float*)d_in[9];
    const float* b1       = (const float*)d_in[10];
    const float* w2       = (const float*)d_in[11];
    const float* b2       = (const float*)d_in[12];
    const float* w3       = (const float*)d_in[13];
    const float* b3       = (const float*)d_in[14];
    float* out = (float*)d_out;

    const int Nn  = in_sizes[2];               // 100000
    const int E   = in_sizes[1] / 2;           // 600000
    const int H   = in_sizes[4];               // 128
    const int Fin = in_sizes[0] / Nn;          // 74
    const int L   = in_sizes[6] / H;           // 3
    const int G   = out_size;                  // 4096
    const int KinPad = ((Fin + 31) / 32) * 32; // 96
    const int H2 = 2 * H;

    const int* src = eidx;
    const int* dst = eidx + E;

    // ---- workspace bump allocator
    char* wsb = (char*)d_ws;
    size_t off = 0;
    auto alloc = [&](size_t bytes) -> void* {
        off = (off + 255) & ~(size_t)255;
        void* p = wsb + off;
        off += bytes;
        return p;
    };
    float*  dinv   = (float*)alloc((size_t)Nn * 4);
    __bf16* x_bf   = (__bf16*)alloc((size_t)Nn * KinPad * 2);
    __bf16* h_bf   = (__bf16*)alloc((size_t)Nn * H * 2);
    float*  hw     = (float*)alloc((size_t)Nn * H * 4);
    float*  hnew   = (float*)alloc((size_t)Nn * H * 4);
    __bf16* embWp  = (__bf16*)alloc((size_t)KinPad * H * 2);
    __bf16* convWp = (__bf16*)alloc((size_t)L * H * H * 2);
    __bf16* w1p    = (__bf16*)alloc((size_t)H2 * H * 2);
    __bf16* w2p    = (__bf16*)alloc((size_t)H * (H / 2) * 2);
    float*  sums   = (float*)alloc((size_t)H * 4);
    float*  sqs    = (float*)alloc((size_t)H * 4);
    float*  scale  = (float*)alloc((size_t)H * 4);
    float*  shift  = (float*)alloc((size_t)H * 4);
    float*  cnt    = (float*)alloc((size_t)G * 4);
    float*  gsum   = (float*)alloc((size_t)G * H * 4);
    unsigned int* gmax = (unsigned int*)alloc((size_t)G * H * 4);
    __bf16* gcat   = (__bf16*)alloc((size_t)G * H2 * 2);
    __bf16* m1_bf  = (__bf16*)alloc((size_t)G * H * 2);
    float*  m2     = (float*)alloc((size_t)G * (H / 2) * 4);
    (void)ws_size; (void)n_in;

    const int B256 = 256;

    // ---- GEMM dispatch: K in {96,128,256} with NT=4, else generic NT=1
    auto gemm = [&](const __bf16* A, int lda, const __bf16* Bp, const float* bias,
                    float* C, __bf16* Cb, int ldc, int M, int N, int K, int relu) {
        int nTilesN = N / 16;
        if ((nTilesN % 4) == 0 && (K == 96 || K == 128 || K == 256)) {
            int tiles = (M / 16) * (nTilesN / 4);
            int blocks = cdiv(tiles, 8);
            if (K == 96)
                k_gemm_wmma<96, 4><<<blocks, B256, 0, stream>>>(A, lda, Bp, bias, C, Cb, ldc, M, N, relu);
            else if (K == 128)
                k_gemm_wmma<128, 4><<<blocks, B256, 0, stream>>>(A, lda, Bp, bias, C, Cb, ldc, M, N, relu);
            else
                k_gemm_wmma<256, 4><<<blocks, B256, 0, stream>>>(A, lda, Bp, bias, C, Cb, ldc, M, N, relu);
        } else {
            int tiles = (M / 16) * nTilesN;
            k_gemm_wmma_gen<<<cdiv(tiles, 8), B256, 0, stream>>>(A, lda, K, Bp, bias, C, Cb, ldc, M, N, relu);
        }
    };

    // ---- degree / normalization
    k_zero_f32<<<cdiv(Nn, B256), B256, 0, stream>>>(dinv, Nn);
    k_deg<<<cdiv(E, B256), B256, 0, stream>>>(dst, E, dinv);
    k_dinv<<<cdiv(Nn, B256), B256, 0, stream>>>(dinv, Nn);

    // ---- pack operands
    k_pack_a<<<cdiv((long)Nn * KinPad, B256), B256, 0, stream>>>(x, Fin, KinPad, x_bf, Nn);
    {
        long t = (long)(KinPad / 32) * (H / 16) * 512;
        k_pack_b<<<cdiv(t, B256), B256, 0, stream>>>(emb_W, Fin, H, KinPad, embWp);
        t = (long)(H / 32) * (H / 16) * 512;
        for (int i = 0; i < L; ++i)
            k_pack_b<<<cdiv(t, B256), B256, 0, stream>>>(conv_W + (long)i * H * H, H, H, H,
                                                         convWp + (long)i * H * H);
        t = (long)(H2 / 32) * (H / 16) * 512;
        k_pack_b<<<cdiv(t, B256), B256, 0, stream>>>(w1, H2, H, H2, w1p);
        t = (long)(H / 32) * ((H / 2) / 16) * 512;
        k_pack_b<<<cdiv(t, B256), B256, 0, stream>>>(w2, H, H / 2, H, w2p);
    }

    // ---- embedding: h_bf = relu(x @ emb_W + emb_b)  (bf16 output only)
    gemm(x_bf, KinPad, embWp, emb_b, nullptr, h_bf, H, Nn, H, KinPad, 1);

    // ---- GCN layers (h_bf carries activations; hnew holds f32 for BN/pooling)
    for (int i = 0; i < L; ++i) {
        gemm(h_bf, H, convWp + (long)i * H * H, nullptr, hw, nullptr, H, Nn, H, H, 0);
        k_init_agg<<<cdiv((long)Nn * H, B256), B256, 0, stream>>>(
            hw, dinv, conv_b + (long)i * H, hnew, Nn, H);
        k_scatter<<<cdiv((long)E * (H / 4), B256), B256, 0, stream>>>(
            src, dst, dinv, hw, hnew, E, H);
        k_zero_f32<<<1, B256, 0, stream>>>(sums, H);
        k_zero_f32<<<1, B256, 0, stream>>>(sqs, H);
        k_bn_partial<<<512, 2 * H, 0, stream>>>(hnew, Nn, H, sums, sqs);
        k_bn_finalize<<<1, H, 0, stream>>>(sums, sqs, bn_gamma + (long)i * H,
                                           bn_beta + (long)i * H, scale, shift, Nn, H);
        k_bn_apply<<<cdiv((long)Nn * H, B256), B256, 0, stream>>>(hnew, h_bf, scale, shift, Nn, H);
    }

    // ---- pooling (hnew holds the final f32 activations)
    k_zero_f32<<<cdiv(G, B256), B256, 0, stream>>>(cnt, G);
    k_zero_f32<<<cdiv((long)G * H, B256), B256, 0, stream>>>(gsum, (long)G * H);
    k_zero_f32<<<cdiv((long)G * H, B256), B256, 0, stream>>>((float*)gmax, (long)G * H);
    k_pool_count<<<cdiv(Nn, B256), B256, 0, stream>>>(batch, Nn, cnt);
    k_pool_accum<<<cdiv((long)Nn * H, B256), B256, 0, stream>>>(batch, hnew, Nn, H, gsum, gmax);
    k_build_g<<<cdiv((long)G * H2, B256), B256, 0, stream>>>(gsum, gmax, cnt, gcat, G, H);

    // ---- MLP head
    gemm(gcat, H2, w1p, b1, nullptr, m1_bf, H, G, H, H2, 1);
    gemm(m1_bf, H, w2p, b2, m2, nullptr, H / 2, G, H / 2, H, 1);
    k_mlp_out<<<cdiv(G, B256), B256, 0, stream>>>(m2, w3, b3, out, G, H / 2);
}
